// SS2D_19061064859763
// MI455X (gfx1250) — matrix operand order repaired and checked by reading, not compiled, not run on hardware
//
#include <hip/hip_runtime.h>
#include <hip/hip_bf16.h>
#include <hip/hip_fp16.h>

// ---------------------------------------------------------------------------
// SS2D (VMamba selective-scan block) for gfx1250 / MI455X.
//  - GEMMs via v_wmma_f32_16x16x32_f16 (fp16 in, fp32 accumulate)
//  - A tile (16 x K, contiguous in memory) staged to LDS per block with
//    global_load_async_to_lds_b128 + s_wait_asynccnt (CDNA5 async path),
//    fragments then read back with ds_load_b128
//  - weights converted+transposed once to [N][K] fp16 so B fragments are
//    contiguous 16B loads per lane (contiguous K order, same as A)
//  - 4-direction selective scan: one (channel,state) per lane, wave32,
//    width-16 shuffle reduction, global_atomic_add_f32 direction merge
// ---------------------------------------------------------------------------

typedef __attribute__((ext_vector_type(16))) _Float16 v16h;
typedef __attribute__((ext_vector_type(8)))  float    v8f;

#define D_MODEL 128
#define D_STATE 16
#define D_INNER 256
#define NB 2
#define NH 64
#define NW 64
#define NTOK (NB*NH*NW)   /* 8192 tokens */

union FragAB { v16h v; uint4 q[2]; };

// ---------------------------------------------------------------------------
// WMMA GEMM:  C[M,N] = A[M,K] (f16, row-major) * Bt[N,K]^T (f16) (+ bias)
// block = 128 threads = 4 waves; block covers 16(M) x 128(N); each wave owns
// two 16x16 N-tiles sharing one A fragment (2 WMMAs per K-step).
// A tile async-staged to LDS once per block.  grid = (N/128, M/16).
// K multiple of 32 (<=256), M mult of 16, N mult of 128.
// ---------------------------------------------------------------------------
__global__ __launch_bounds__(128)
void wmma_gemm_kernel(const _Float16* __restrict__ A,
                      const _Float16* __restrict__ Bt,
                      float* __restrict__ C,
                      const float* __restrict__ bias,
                      int M, int N, int K)
{
    __shared__ _Float16 Atile[16 * 256];      // up to 8 KB of the 320KB/WGP LDS

    const int tid  = threadIdx.x;
    const int lane = tid & 31;
    const int wave = tid >> 5;
    const int m0 = blockIdx.y * 16;
    const int n0 = blockIdx.x * 128 + wave * 32;

    // ---- async stage of the contiguous 16 x K A tile (32*K bytes) ----------
    {
        const unsigned nchunk = 2u * (unsigned)K;                 // 16B chunks
        const unsigned long long gbase =
            (unsigned long long)(const void*)(A + (size_t)m0 * K);
        for (unsigned c = tid; c < nchunk; c += 128u) {
            const unsigned lds_off =
                (unsigned)(size_t)(&Atile[c * 8]);   // low 32b of generic = LDS byte offset
            const unsigned long long gaddr = gbase + (unsigned long long)c * 16ull;
            asm volatile("global_load_async_to_lds_b128 %0, %1, off"
                         :: "v"(lds_off), "v"(gaddr) : "memory");
        }
        asm volatile("s_wait_asynccnt 0x0" ::: "memory");
        __syncthreads();
    }

    const int hi = lane >> 4;   // 0..1 (half-wave select)
    const int lo = lane & 15;   // 0..15

    // A fragment (from LDS): lane (lo+16hi) holds row lo, K = [16hi .. 16hi+15]
    const _Float16* As = &Atile[lo * K];
    // B fragments: lane (lo+16hi) holds col n, K = [16hi .. 16hi+15]
    const _Float16* B0 = Bt + (size_t)(n0 + lo) * K;
    const _Float16* B1 = Bt + (size_t)(n0 + 16 + lo) * K;

    v8f acc0 = {};
    v8f acc1 = {};
    for (int k0 = 0; k0 < K; k0 += 32) {
        FragAB a, b0, b1;
        a.q[0]  = *reinterpret_cast<const uint4*>(As + k0 + 16 * hi);      // ds_load_b128
        a.q[1]  = *reinterpret_cast<const uint4*>(As + k0 + 16 * hi + 8);
        b0.q[0] = *reinterpret_cast<const uint4*>(B0 + k0 + 16 * hi);
        b0.q[1] = *reinterpret_cast<const uint4*>(B0 + k0 + 16 * hi + 8);
        b1.q[0] = *reinterpret_cast<const uint4*>(B1 + k0 + 16 * hi);
        b1.q[1] = *reinterpret_cast<const uint4*>(B1 + k0 + 16 * hi + 8);
        if (k0 + 32 < K) {
            __builtin_prefetch(B0 + k0 + 32, 0, 0);   // global_prefetch_b8
            __builtin_prefetch(B1 + k0 + 32, 0, 0);
        }
        acc0 = __builtin_amdgcn_wmma_f32_16x16x32_f16(false, a.v, false, b0.v,
                                                      (short)0, acc0, false, false);
        acc1 = __builtin_amdgcn_wmma_f32_16x16x32_f16(false, a.v, false, b1.v,
                                                      (short)0, acc1, false, false);
    }

    const int nc0 = n0 + lo;
    const int nc1 = n0 + 16 + lo;
    const float bv0 = bias ? bias[nc0] : 0.0f;
    const float bv1 = bias ? bias[nc1] : 0.0f;
    #pragma unroll
    for (int r = 0; r < 8; ++r) {
        const int mr = m0 + r + 8 * hi;          // C/D layout: M = r + 8*hi, N = lo
        C[(size_t)mr * N + nc0] = acc0[r] + bv0;
        C[(size_t)mr * N + nc1] = acc1[r] + bv1;
    }
}

// ---------------------------------------------------------------------------
// Utility kernels
// ---------------------------------------------------------------------------
__global__ __launch_bounds__(256)
void f32_to_f16_kernel(const float* __restrict__ in, _Float16* __restrict__ out, int n)
{
    int i = blockIdx.x * 256 + threadIdx.x;
    if (i < n) out[i] = (_Float16)in[i];
}

__global__ __launch_bounds__(256)
void transpose_f32_to_f16_kernel(const float* __restrict__ in,  // [K][N]
                                 _Float16* __restrict__ out,    // [N][K]
                                 int K, int N)
{
    int i = blockIdx.x * 256 + threadIdx.x;
    if (i < K * N) {
        int k = i / N, n = i - k * N;
        out[(size_t)n * K + k] = (_Float16)in[i];
    }
}

__global__ __launch_bounds__(256)
void zero_f32_kernel(float* __restrict__ p, int n)
{
    int i = blockIdx.x * 256 + threadIdx.x;
    if (i < n) p[i] = 0.0f;
}

// ---------------------------------------------------------------------------
// Depthwise 3x3 conv (SAME) + bias + SiLU.  block = one token, thread = channel.
// Reads x_in = xz[:, 0:256]; writes f32 (scan input) and f16 (GEMM input).
// ---------------------------------------------------------------------------
__global__ __launch_bounds__(256)
void conv_silu_kernel(const float* __restrict__ xz,
                      const float* __restrict__ conv_w,   // [3][3][1][256]
                      const float* __restrict__ conv_b,   // [256]
                      float* __restrict__ xc,
                      _Float16* __restrict__ xch)
{
    const int tok = blockIdx.x;
    const int c   = threadIdx.x;
    const int b = tok >> 12, h = (tok >> 6) & 63, w = tok & 63;

    float acc = conv_b[c];
    #pragma unroll
    for (int kh = 0; kh < 3; ++kh) {
        const int hh = h + kh - 1;
        if ((unsigned)hh >= (unsigned)NH) continue;
        #pragma unroll
        for (int kw = 0; kw < 3; ++kw) {
            const int ww = w + kw - 1;
            if ((unsigned)ww >= (unsigned)NW) continue;
            const int t2 = (b << 12) + (hh << 6) + ww;
            acc += conv_w[(kh * 3 + kw) * D_INNER + c] * xz[(size_t)t2 * 512 + c];
        }
    }
    const float s = acc / (1.0f + __expf(-acc));   // SiLU
    xc [(size_t)tok * D_INNER + c] = s;
    xch[(size_t)tok * D_INNER + c] = (_Float16)s;
}

// ---------------------------------------------------------------------------
// dt_raw = mean over the 4 copies in x_dbl[:, 0:512] -> f16 for dt GEMM
// ---------------------------------------------------------------------------
__global__ __launch_bounds__(128)
void dtraw_kernel(const float* __restrict__ xdbl, _Float16* __restrict__ dtraw_h)
{
    const int tok = blockIdx.x;
    const int c   = threadIdx.x;                 // 0..127
    const float* p = xdbl + (size_t)tok * 640;
    const float v = 0.25f * (p[c] + p[128 + c] + p[256 + c] + p[384 + c]);
    dtraw_h[(size_t)tok * D_MODEL + c] = (_Float16)v;
}

// ---------------------------------------------------------------------------
// 4-direction selective scan.
// wave32: lane = 16*dp + n, dp in {0,1} (channel-pair half), n = state.
// wave id -> (batch, dir, channel pair).  1024 waves total, 8 waves/block.
// Recurrence: s = exp(delta*A)*s + delta*u*B ;  y = sum_n s*C + u*D
// Directions merged with 0.25-weighted global_atomic_add_f32.
// ---------------------------------------------------------------------------
__global__ __launch_bounds__(256)
void scan_kernel(const float* __restrict__ dt,      // [8192][1024]
                 const float* __restrict__ xc,      // [8192][256]
                 const float* __restrict__ xdbl,    // [8192][640] (B at 512, C at 576)
                 const float* __restrict__ A_log,   // [256][16]
                 const float* __restrict__ D_param, // [256]
                 float* __restrict__ yacc)          // [8192][256]
{
    const int wid  = blockIdx.x * 8 + (threadIdx.x >> 5);   // 0..1023
    const int lane = threadIdx.x & 31;
    const int n  = lane & 15;
    const int dp = lane >> 4;
    const int cpair = wid & 127;
    const int dir   = (wid >> 7) & 3;
    const int b     = wid >> 9;
    const int d = cpair * 2 + dp;

    const float A_dn = -__expf(A_log[d * D_STATE + n]);
    const float Dd   = D_param[d];
    float s = 0.0f;

    for (int t = 0; t < NH * NW; ++t) {
        const int a0 = t >> 6, a1 = t & 63;
        int h, w;
        if      (dir == 0) { h = a0;      w = a1;      }  // raster
        else if (dir == 1) { h = a0;      w = 63 - a1; }  // flip W
        else if (dir == 2) { h = a1;      w = a0;      }  // transpose
        else               { h = 63 - a1; w = a0;      }  // flip H + transpose
        const int tok = (b << 12) + (h << 6) + w;

        const float delta = dt[(size_t)tok * 1024 + dir * D_INNER + d];
        const float u     = xc[(size_t)tok * D_INNER + d];
        const float* pbc  = xdbl + (size_t)tok * 640;
        const float Bv = pbc[512 + dir * D_STATE + n];
        const float Cv = pbc[576 + dir * D_STATE + n];

        s = __expf(delta * A_dn) * s + (delta * u) * Bv;

        float p = s * Cv;                 // reduce over 16 states
        p += __shfl_xor(p, 1, 16);
        p += __shfl_xor(p, 2, 16);
        p += __shfl_xor(p, 4, 16);
        p += __shfl_xor(p, 8, 16);
        if (n == 0) {
            atomicAdd(&yacc[(size_t)tok * D_INNER + d], 0.25f * (p + u * Dd));
        }
    }
}

// ---------------------------------------------------------------------------
// Gate: g = y_avg * silu(z)  ->  f16 for the output GEMM
// ---------------------------------------------------------------------------
__global__ __launch_bounds__(256)
void gate_kernel(const float* __restrict__ yacc,
                 const float* __restrict__ xz,      // z at [:, 256:512]
                 _Float16* __restrict__ gh)
{
    const int tok = blockIdx.x;
    const int c   = threadIdx.x;
    const float z = xz[(size_t)tok * 512 + 256 + c];
    const float g = yacc[(size_t)tok * D_INNER + c] * (z / (1.0f + __expf(-z)));
    gh[(size_t)tok * D_INNER + c] = (_Float16)g;
}

// ---------------------------------------------------------------------------
// Workspace layout (bytes, each region 256B aligned by construction)
// ---------------------------------------------------------------------------
static constexpr size_t O_XH     = 0;                          // 8192*128 f16
static constexpr size_t O_WTIN   = O_XH     + 2097152;         // 512*128 f16
static constexpr size_t O_WTXP   = O_WTIN   + 131072;          // 640*256 f16
static constexpr size_t O_WTDT   = O_WTXP   + 327680;          // 1024*128 f16
static constexpr size_t O_WTOUT  = O_WTDT   + 262144;          // 128*256 f16
static constexpr size_t O_XZ     = O_WTOUT  + 65536;           // 8192*512 f32
static constexpr size_t O_XC     = O_XZ     + 16777216;        // 8192*256 f32
static constexpr size_t O_XCH    = O_XC     + 8388608;         // 8192*256 f16
static constexpr size_t O_XDBL   = O_XCH    + 4194304;         // 8192*640 f32
static constexpr size_t O_DTRAW  = O_XDBL   + 20971520;        // 8192*128 f16
static constexpr size_t O_DT     = O_DTRAW  + 2097152;         // 8192*1024 f32
static constexpr size_t O_YACC   = O_DT     + 33554432;        // 8192*256 f32
static constexpr size_t O_GH     = O_YACC   + 8388608;         // 8192*256 f16

extern "C" void kernel_launch(void* const* d_in, const int* in_sizes, int n_in,
                              void* d_out, int out_size, void* d_ws, size_t ws_size,
                              hipStream_t stream)
{
    const float* x       = (const float*)d_in[0];  // [2,64,64,128]
    const float* W_in    = (const float*)d_in[1];  // [128,512]
    const float* conv_w  = (const float*)d_in[2];  // [3,3,1,256]
    const float* conv_b  = (const float*)d_in[3];  // [256]
    const float* W_xproj = (const float*)d_in[4];  // [256,640]
    const float* W_dt    = (const float*)d_in[5];  // [128,1024]
    const float* b_dt    = (const float*)d_in[6];  // [1024]
    const float* A_log   = (const float*)d_in[7];  // [256,16]
    const float* D_param = (const float*)d_in[8];  // [256]
    const float* W_out   = (const float*)d_in[9];  // [256,128]

    char* ws = (char*)d_ws;
    _Float16* xh      = (_Float16*)(ws + O_XH);
    _Float16* Wt_in   = (_Float16*)(ws + O_WTIN);
    _Float16* Wt_xp   = (_Float16*)(ws + O_WTXP);
    _Float16* Wt_dt   = (_Float16*)(ws + O_WTDT);
    _Float16* Wt_out  = (_Float16*)(ws + O_WTOUT);
    float*    xz      = (float*)   (ws + O_XZ);
    float*    xc      = (float*)   (ws + O_XC);
    _Float16* xch     = (_Float16*)(ws + O_XCH);
    float*    xdbl    = (float*)   (ws + O_XDBL);
    _Float16* dtraw_h = (_Float16*)(ws + O_DTRAW);
    float*    dt      = (float*)   (ws + O_DT);
    float*    yacc    = (float*)   (ws + O_YACC);
    _Float16* gh      = (_Float16*)(ws + O_GH);

    float* out = (float*)d_out;    // [8192][128] f32

    // --- stage 0: precision conversion + weight transposes, zero accumulators
    f32_to_f16_kernel<<<(NTOK * D_MODEL + 255) / 256, 256, 0, stream>>>(x, xh, NTOK * D_MODEL);
    transpose_f32_to_f16_kernel<<<(128 * 512  + 255) / 256, 256, 0, stream>>>(W_in,    Wt_in,  128, 512);
    transpose_f32_to_f16_kernel<<<(256 * 640  + 255) / 256, 256, 0, stream>>>(W_xproj, Wt_xp,  256, 640);
    transpose_f32_to_f16_kernel<<<(128 * 1024 + 255) / 256, 256, 0, stream>>>(W_dt,    Wt_dt,  128, 1024);
    transpose_f32_to_f16_kernel<<<(256 * 128  + 255) / 256, 256, 0, stream>>>(W_out,   Wt_out, 256, 128);
    zero_f32_kernel<<<(NTOK * D_INNER + 255) / 256, 256, 0, stream>>>(yacc, NTOK * D_INNER);

    // --- stage 1: xz = x @ W_in   (8192 x 512, K=128)
    wmma_gemm_kernel<<<dim3(512 / 128, NTOK / 16), 128, 0, stream>>>(
        xh, Wt_in, xz, nullptr, NTOK, 512, 128);

    // --- stage 2: depthwise conv + SiLU
    conv_silu_kernel<<<NTOK, 256, 0, stream>>>(xz, conv_w, conv_b, xc, xch);

    // --- stage 3: x_dbl = x_in @ W_xproj   (8192 x 640, K=256)
    wmma_gemm_kernel<<<dim3(640 / 128, NTOK / 16), 128, 0, stream>>>(
        xch, Wt_xp, xdbl, nullptr, NTOK, 640, 256);

    // --- stage 4: dt_raw mean, dt = dt_raw @ W_dt + b_dt  (8192 x 1024, K=128)
    dtraw_kernel<<<NTOK, 128, 0, stream>>>(xdbl, dtraw_h);
    wmma_gemm_kernel<<<dim3(1024 / 128, NTOK / 16), 128, 0, stream>>>(
        dtraw_h, Wt_dt, dt, b_dt, NTOK, 1024, 128);

    // --- stage 5: 4-direction selective scan (1024 waves = 128 blocks x 8 waves)
    scan_kernel<<<128, 256, 0, stream>>>(dt, xc, xdbl, A_log, D_param, yacc);

    // --- stage 6: gate + output GEMM  (8192 x 128, K=256)
    gate_kernel<<<NTOK, 256, 0, stream>>>(yacc, xz, gh);
    wmma_gemm_kernel<<<dim3(128 / 128, NTOK / 16), 128, 0, stream>>>(
        gh, Wt_out, out, nullptr, NTOK, 128, 256);
}
